// Attention_36799279793005
// MI455X (gfx1250) — compile-verified, matrix operand order checked by reading, not run on hardware
//
#include <hip/hip_runtime.h>

// ---------------------------------------------------------------------------
// Problem dims (compile-time, from reference)
// ---------------------------------------------------------------------------
#define BATCH 2
#define SEQ   2048
#define DIM   2048
#define HEADS 32
#define DH    64
#define BH    (BATCH * HEADS)        // 64
#define MROWS (BATCH * SEQ)          // 4096  (flattened b,s)
#define THETA 10000.0f

// ---------------------------------------------------------------------------
// CDNA5 WMMA types / helpers (wave32)
// ---------------------------------------------------------------------------
typedef unsigned int u32;
typedef __attribute__((ext_vector_type(4)))  u32    v4u;
typedef __attribute__((ext_vector_type(8)))  float  v8f;
typedef __attribute__((ext_vector_type(16))) __bf16 v16bf;

union Frag { v16bf v; v4u q[2]; };

// D = A(16x32 bf16) x B(32x16 bf16) + C(16x16 f32)
#define WMMA_BF16(A, B, C) \
  __builtin_amdgcn_wmma_f32_16x16x32_bf16(false, (A), false, (B), (short)0, (C), false, false)

// Load one 16x32 bf16 fragment (A layout; B uses the same lane pattern with
// the "row" being the N column) from row-major memory.
//   lane r = lane&15 -> matrix row (A) / column (B)
//   half h = lane>>4 -> elems 0..7  = K (k0 + h*8 .. +7)        (16B chunk)
//                       elems 8..15 = K (k0 + 16 + h*8 .. +7)   (16B chunk)
__device__ __forceinline__ v16bf load_frag(const __bf16* __restrict__ base,
                                           int ld, int k0, int lane) {
  const int rr = lane & 15;
  const int hh = lane >> 4;
  const __bf16* p = base + (size_t)rr * (size_t)ld;
  Frag f;
  f.q[0] = *(const v4u*)(p + k0 + hh * 8);
  f.q[1] = *(const v4u*)(p + k0 + 16 + hh * 8);
  return f.v;
}

__device__ __forceinline__ unsigned short f2bf(float f) {
  __bf16 h = (__bf16)f;
  return __builtin_bit_cast(unsigned short, h);
}

// ---------------------------------------------------------------------------
// 1) fp32 -> bf16 elementwise convert
// ---------------------------------------------------------------------------
__global__ __launch_bounds__(256) void cvt_bf16_kernel(const float* __restrict__ src,
                                                       __bf16* __restrict__ dst, int n) {
  int i = blockIdx.x * 256 + threadIdx.x;
  if (i < n) dst[i] = (__bf16)src[i];
}

// ---------------------------------------------------------------------------
// 2) Row l2-norm of a [rows x cols] fp32 matrix -> bf16 (wave per row)
// ---------------------------------------------------------------------------
__global__ __launch_bounds__(128) void wnorm_row_kernel(const float* __restrict__ w,
                                                        __bf16* __restrict__ out, int cols) {
  int row  = (blockIdx.x * blockDim.x + threadIdx.x) >> 5;
  int lane = threadIdx.x & 31;
  const float* r = w + (size_t)row * cols;
  float ss = 0.f;
  for (int i = lane; i < cols; i += 32) { float v = r[i]; ss += v * v; }
  #pragma unroll
  for (int off = 16; off > 0; off >>= 1) ss += __shfl_xor(ss, off, 32);
  float inv = 1.0f / fmaxf(sqrtf(ss), 1e-12f);
  for (int i = lane; i < cols; i += 32) out[(size_t)row * cols + i] = (__bf16)(r[i] * inv);
}

// ---------------------------------------------------------------------------
// 3) Column l2-norm of wo [DIM x DIM] (axis 0) -> bf16 (wave per column)
// ---------------------------------------------------------------------------
__global__ __launch_bounds__(128) void wnorm_col_kernel(const float* __restrict__ w,
                                                        __bf16* __restrict__ out) {
  int col  = (blockIdx.x * blockDim.x + threadIdx.x) >> 5;
  int lane = threadIdx.x & 31;
  float ss = 0.f;
  for (int d = lane; d < DIM; d += 32) { float v = w[(size_t)d * DIM + col]; ss += v * v; }
  #pragma unroll
  for (int off = 16; off > 0; off >>= 1) ss += __shfl_xor(ss, off, 32);
  float inv = 1.0f / fmaxf(sqrtf(ss), 1e-12f);
  for (int d = lane; d < DIM; d += 32)
    out[(size_t)d * DIM + col] = (__bf16)(w[(size_t)d * DIM + col] * inv);
}

// ---------------------------------------------------------------------------
// 4) Projection GEMM: Y[r,e] = sum_d Xb[r,d] * Wb[e,d]   (both row-major bf16)
//    Wave computes a 64x32 tile (4 M-subtiles x 2 N-subtiles):
//      8 WMMAs per 12 x b128 fragment loads per k-step.
//    f32 result scattered to [b,h,s,dh].
// ---------------------------------------------------------------------------
__global__ __launch_bounds__(128) void gemm_proj_kernel(const __bf16* __restrict__ A,
                                                        const __bf16* __restrict__ Bw,
                                                        float* __restrict__ dst) {
  int lane = threadIdx.x & 31;
  int wid  = blockIdx.x * 4 + (threadIdx.x >> 5);
  int m0 = (wid >> 6) << 6;     // 64 M-tiles of 64
  int n0 = (wid & 63) << 5;     // 64 N-tiles of 32
  v8f acc[4][2] = {};
  for (int k0 = 0; k0 < DIM; k0 += 32) {
    v16bf b0 = load_frag(Bw + (size_t)n0 * DIM,        DIM, k0, lane);
    v16bf b1 = load_frag(Bw + (size_t)(n0 + 16) * DIM, DIM, k0, lane);
    #pragma unroll
    for (int mi = 0; mi < 4; mi++) {
      v16bf a = load_frag(A + (size_t)(m0 + mi * 16) * DIM, DIM, k0, lane);
      acc[mi][0] = WMMA_BF16(a, b0, acc[mi][0]);
      acc[mi][1] = WMMA_BF16(a, b1, acc[mi][1]);
    }
  }
  int rr = lane & 15, hh = lane >> 4;
  #pragma unroll
  for (int mi = 0; mi < 4; mi++)
    #pragma unroll
    for (int ni = 0; ni < 2; ni++)
      #pragma unroll
      for (int r = 0; r < 8; r++) {
        int R = m0 + mi * 16 + r + 8 * hh;     // flattened (b,s)
        int E = n0 + ni * 16 + rr;             // flattened (h,dh)
        int b = R >> 11, s = R & (SEQ - 1);
        int head = E >> 6, dh = E & (DH - 1);
        dst[(((size_t)(b * HEADS + head)) * SEQ + s) * DH + dh] = acc[mi][ni][r];
      }
}

// ---------------------------------------------------------------------------
// 5) Per-(b,h,s) head row: l2norm * qk_scale, then RoPE; f32 -> bf16.
//    Wave per row; lane owns the rotation pair (2i, 2i+1).
// ---------------------------------------------------------------------------
__global__ __launch_bounds__(128) void rope_norm_kernel(const float* __restrict__ src,
                                                        const float* __restrict__ qk_scale,
                                                        __bf16* __restrict__ dst) {
  int row  = (blockIdx.x * blockDim.x + threadIdx.x) >> 5;   // [0, BH*SEQ)
  int lane = threadIdx.x & 31;
  int spos = row & (SEQ - 1);
  const float* r = src + (size_t)row * DH;
  float v0 = r[2 * lane], v1 = r[2 * lane + 1];
  float ss = v0 * v0 + v1 * v1;
  #pragma unroll
  for (int off = 16; off > 0; off >>= 1) ss += __shfl_xor(ss, off, 32);
  float inv = 1.0f / fmaxf(sqrtf(ss), 1e-12f);
  float t0 = v0 * inv * qk_scale[2 * lane];
  float t1 = v1 * inv * qk_scale[2 * lane + 1];
  float freq = __powf(THETA, -(float)lane / 32.0f);   // theta^(-(2i)/DH)
  float ang = (float)spos * freq;
  float c, sn;
  __sincosf(ang, &sn, &c);
  dst[(size_t)row * DH + 2 * lane]     = (__bf16)(t0 * c - t1 * sn);
  dst[(size_t)row * DH + 2 * lane + 1] = (__bf16)(t1 * c + t0 * sn);
}

// ---------------------------------------------------------------------------
// 6) V relayout: f32 [bh,s,dh] -> bf16 [bh,dh,s] (so PV B-fragments are contiguous)
// ---------------------------------------------------------------------------
__global__ __launch_bounds__(256) void v_relayout_kernel(const float* __restrict__ vf,
                                                         __bf16* __restrict__ vt) {
  int i = blockIdx.x * 256 + threadIdx.x;            // BH*SEQ*DH = 8388608 exact
  int dh = i & (DH - 1);
  int s  = (i >> 6) & (SEQ - 1);
  int bh = i >> 17;                                   // SEQ*DH = 2^17
  vt[((size_t)bh * DH + dh) * SEQ + s] = (__bf16)vf[i];
}

// ---------------------------------------------------------------------------
// 7) Flash attention (causal, online softmax). Wave per 16-row query tile.
//    Per 32-key tile: 4 WMMAs (logits) + LDS reshape of P + 4 WMMAs (P*V).
// ---------------------------------------------------------------------------
__global__ __launch_bounds__(128) void flash_kernel(const __bf16* __restrict__ qb,
                                                    const __bf16* __restrict__ kb,
                                                    const __bf16* __restrict__ vt,
                                                    __bf16* __restrict__ ob) {
  __shared__ unsigned short sP[4 * 16 * 32];          // 1KB per wave
  int lane = threadIdx.x & 31;
  int wl   = threadIdx.x >> 5;
  int wid  = blockIdx.x * 4 + wl;
  int bh = wid >> 7;                                  // SEQ/16 = 128 q-tiles / head
  int q0 = (wid & 127) << 4;
  int rr = lane & 15, hh = lane >> 4;
  unsigned short* myP = sP + wl * 512;

  const __bf16* qbase = qb + ((size_t)bh * SEQ + q0) * DH;
  v16bf aq0 = load_frag(qbase, DH, 0,  lane);
  v16bf aq1 = load_frag(qbase, DH, 32, lane);

  const __bf16* kbase = kb + (size_t)bh * SEQ * DH;
  const __bf16* vbase = vt + (size_t)bh * DH * SEQ;

  v8f o0 = {}, o1 = {}, o2 = {}, o3 = {};
  float mrun[8], lrun[8];
  #pragma unroll
  for (int r = 0; r < 8; r++) { mrun[r] = -1e38f; lrun[r] = 0.f; }

  for (int j0 = 0; j0 <= q0 + 15; j0 += 32) {
    // ---- logits tile S[16 x 32] = Q(16x64) . K(j0..j0+31)^T -------------
    v16bf bk0lo = load_frag(kbase + (size_t)j0 * DH,        DH, 0,  lane);
    v16bf bk0hi = load_frag(kbase + (size_t)j0 * DH,        DH, 32, lane);
    v16bf bk1lo = load_frag(kbase + (size_t)(j0 + 16) * DH, DH, 0,  lane);
    v16bf bk1hi = load_frag(kbase + (size_t)(j0 + 16) * DH, DH, 32, lane);
    v8f s0 = {}, s1 = {};
    s0 = WMMA_BF16(aq0, bk0lo, s0);
    s0 = WMMA_BF16(aq1, bk0hi, s0);
    s1 = WMMA_BF16(aq0, bk1lo, s1);
    s1 = WMMA_BF16(aq1, bk1hi, s1);

    if (j0 + 31 > q0) {                                 // causal mask (wave-uniform)
      #pragma unroll
      for (int r = 0; r < 8; r++) {
        int row = q0 + r + 8 * hh;
        if (j0 + rr      > row) s0[r] = -3e38f;
        if (j0 + 16 + rr > row) s1[r] = -3e38f;
      }
    }

    // ---- online softmax row stats (rows live per-lane in C layout) ------
    float alpha[8];
    #pragma unroll
    for (int r = 0; r < 8; r++) {
      float t = fmaxf(s0[r], s1[r]);
      t = fmaxf(t, __shfl_xor(t, 1, 32));
      t = fmaxf(t, __shfl_xor(t, 2, 32));
      t = fmaxf(t, __shfl_xor(t, 4, 32));
      t = fmaxf(t, __shfl_xor(t, 8, 32));
      float mnew = fmaxf(mrun[r], t);
      alpha[r] = __expf(mrun[r] - mnew);
      float p0 = __expf(s0[r] - mnew);
      float p1 = __expf(s1[r] - mnew);
      s0[r] = p0; s1[r] = p1;
      float ts = p0 + p1;
      ts += __shfl_xor(ts, 1, 32);
      ts += __shfl_xor(ts, 2, 32);
      ts += __shfl_xor(ts, 4, 32);
      ts += __shfl_xor(ts, 8, 32);
      lrun[r] = lrun[r] * alpha[r] + ts;
      mrun[r] = mnew;
    }

    // ---- rescale O, stage P (bf16) into this wave's LDS tile ------------
    #pragma unroll
    for (int r = 0; r < 8; r++) {
      o0[r] *= alpha[r]; o1[r] *= alpha[r]; o2[r] *= alpha[r]; o3[r] *= alpha[r];
      int rowl = r + 8 * hh;
      myP[rowl * 32 + rr]      = f2bf(s0[r]);
      myP[rowl * 32 + 16 + rr] = f2bf(s1[r]);
    }
    // same-wave DS ops are in order; compiler inserts s_wait_dscnt for the reload
    v16bf pA;
    {
      Frag f;
      f.q[0] = *(const v4u*)(myP + rr * 32 + hh * 8);
      f.q[1] = *(const v4u*)(myP + rr * 32 + 16 + hh * 8);
      pA = f.v;
    }

    // ---- O(16x64) += P(16x32) . V(32x64) --------------------------------
    v16bf bv0 = load_frag(vbase + (size_t)0  * SEQ + j0, SEQ, 0, lane);
    v16bf bv1 = load_frag(vbase + (size_t)16 * SEQ + j0, SEQ, 0, lane);
    v16bf bv2 = load_frag(vbase + (size_t)32 * SEQ + j0, SEQ, 0, lane);
    v16bf bv3 = load_frag(vbase + (size_t)48 * SEQ + j0, SEQ, 0, lane);
    o0 = WMMA_BF16(pA, bv0, o0);
    o1 = WMMA_BF16(pA, bv1, o1);
    o2 = WMMA_BF16(pA, bv2, o2);
    o3 = WMMA_BF16(pA, bv3, o3);
  }

  // ---- epilogue: O /= l, write bf16 to [b*SEQ+s, h*DH+dh] ----------------
  int b = bh >> 5, head = bh & (HEADS - 1);
  #pragma unroll
  for (int r = 0; r < 8; r++) {
    float inv = 1.0f / lrun[r];
    int srow = q0 + r + 8 * hh;
    size_t base = ((size_t)b * SEQ + srow) * DIM + head * DH;
    ob[base + rr]      = (__bf16)(o0[r] * inv);
    ob[base + 16 + rr] = (__bf16)(o1[r] * inv);
    ob[base + 32 + rr] = (__bf16)(o2[r] * inv);
    ob[base + 48 + rr] = (__bf16)(o3[r] * inv);
  }
}

// ---------------------------------------------------------------------------
// 8) Output GEMM: out[r,d] = sum_e Ob[r,e] * won[d,e]  -> fp32 d_out
//    Same 64x32 wave tile as the projection GEMM.
// ---------------------------------------------------------------------------
__global__ __launch_bounds__(128) void gemm_out_kernel(const __bf16* __restrict__ A,
                                                       const __bf16* __restrict__ Bw,
                                                       float* __restrict__ dst) {
  int lane = threadIdx.x & 31;
  int wid  = blockIdx.x * 4 + (threadIdx.x >> 5);
  int m0 = (wid >> 6) << 6;
  int n0 = (wid & 63) << 5;
  v8f acc[4][2] = {};
  for (int k0 = 0; k0 < DIM; k0 += 32) {
    v16bf b0 = load_frag(Bw + (size_t)n0 * DIM,        DIM, k0, lane);
    v16bf b1 = load_frag(Bw + (size_t)(n0 + 16) * DIM, DIM, k0, lane);
    #pragma unroll
    for (int mi = 0; mi < 4; mi++) {
      v16bf a = load_frag(A + (size_t)(m0 + mi * 16) * DIM, DIM, k0, lane);
      acc[mi][0] = WMMA_BF16(a, b0, acc[mi][0]);
      acc[mi][1] = WMMA_BF16(a, b1, acc[mi][1]);
    }
  }
  int rr = lane & 15, hh = lane >> 4;
  #pragma unroll
  for (int mi = 0; mi < 4; mi++)
    #pragma unroll
    for (int ni = 0; ni < 2; ni++)
      #pragma unroll
      for (int r = 0; r < 8; r++) {
        int R = m0 + mi * 16 + r + 8 * hh;
        int E = n0 + ni * 16 + rr;
        dst[(size_t)R * DIM + E] = acc[mi][ni][r];
      }
}

// ---------------------------------------------------------------------------
// Launch: pipeline on `stream`, scratch carved from d_ws (~218 MB used).
// ---------------------------------------------------------------------------
extern "C" void kernel_launch(void* const* d_in, const int* in_sizes, int n_in,
                              void* d_out, int out_size, void* d_ws, size_t ws_size,
                              hipStream_t stream) {
  (void)in_sizes; (void)n_in; (void)out_size; (void)ws_size;
  const float* x  = (const float*)d_in[0];
  const float* wq = (const float*)d_in[1];
  const float* wk = (const float*)d_in[2];
  const float* wv = (const float*)d_in[3];
  const float* wo = (const float*)d_in[4];
  const float* qs = (const float*)d_in[5];

  char* ws = (char*)d_ws;
  size_t off = 0;
  auto take = [&](size_t bytes) -> char* {
    char* p = ws + off;
    off += (bytes + 255) & ~(size_t)255;
    return p;
  };
  const size_t NE = (size_t)MROWS * DIM;              // 8388608 elements
  __bf16* xb  = (__bf16*)take(NE * 2);                // x in bf16
  __bf16* wqb = (__bf16*)take((size_t)DIM * DIM * 2); // normalized weights, bf16
  __bf16* wkb = (__bf16*)take((size_t)DIM * DIM * 2);
  __bf16* wvb = (__bf16*)take((size_t)DIM * DIM * 2);
  __bf16* wob = (__bf16*)take((size_t)DIM * DIM * 2);
  float*  qf  = (float*) take(NE * 4);                // f32 [bh,s,dh]
  float*  kf  = (float*) take(NE * 4);
  float*  vf  = (float*) take(NE * 4);
  __bf16* qbb = (__bf16*)take(NE * 2);                // post-rope bf16 [bh,s,dh]
  __bf16* kbb = (__bf16*)take(NE * 2);
  __bf16* vtb = (__bf16*)take(NE * 2);                // bf16 [bh,dh,s]
  __bf16* obb = (__bf16*)take(NE * 2);                // attention out bf16 [b*s, h*dh]

  // 1) x -> bf16
  cvt_bf16_kernel<<<(int)(NE / 256), 256, 0, stream>>>(x, xb, (int)NE);
  // 2) weight norms
  wnorm_row_kernel<<<DIM / 4, 128, 0, stream>>>(wq, wqb, DIM);
  wnorm_row_kernel<<<DIM / 4, 128, 0, stream>>>(wk, wkb, DIM);
  wnorm_row_kernel<<<DIM / 4, 128, 0, stream>>>(wv, wvb, DIM);
  wnorm_col_kernel<<<DIM / 4, 128, 0, stream>>>(wo, wob);
  // 3) projections (WMMA): 64x32 wave tiles -> 4096 tiles -> 1024 blocks
  gemm_proj_kernel<<<1024, 128, 0, stream>>>(xb, wqb, qf);
  gemm_proj_kernel<<<1024, 128, 0, stream>>>(xb, wkb, kf);
  gemm_proj_kernel<<<1024, 128, 0, stream>>>(xb, wvb, vf);
  // 4) q/k: l2norm * qk_scale + RoPE -> bf16 ; v: relayout -> bf16 [bh,dh,s]
  rope_norm_kernel<<<(BH * SEQ) / 4, 128, 0, stream>>>(qf, qs, qbb);
  rope_norm_kernel<<<(BH * SEQ) / 4, 128, 0, stream>>>(kf, qs, kbb);
  v_relayout_kernel<<<(int)(NE / 256), 256, 0, stream>>>(vf, vtb);
  // 5) causal flash attention (WMMA): BH * (SEQ/16) = 8192 wave-tiles
  flash_kernel<<<2048, 128, 0, stream>>>(qbb, kbb, vtb, obb);
  // 6) output projection (WMMA) -> fp32 d_out
  gemm_out_kernel<<<1024, 128, 0, stream>>>(obb, wob, (float*)d_out);
}
